// MABnoSoftmaxNonNeg_79559974191364
// MI455X (gfx1250) — compile-verified, matrix operand order checked by reading, not run on hardware
//
#include <hip/hip_runtime.h>
#include <hip/hip_bf16.h>

typedef __attribute__((ext_vector_type(16))) __bf16 v16bf;
typedef __attribute__((ext_vector_type(8)))  __bf16 v8bf;
typedef __attribute__((ext_vector_type(4)))  __bf16 v4bf;
typedef __attribute__((ext_vector_type(8)))  float  v8f;

#define WMMA_BF16(a, b, c) \
    __builtin_amdgcn_wmma_f32_16x16x32_bf16(false, (a), false, (b), (short)0, (c), false, false)

// ---------- helpers ----------
__device__ __forceinline__ v4bf f4tobf4(float4 v) {
    v4bf r;
    r[0] = (__bf16)v.x; r[1] = (__bf16)v.y; r[2] = (__bf16)v.z; r[3] = (__bf16)v.w;
    return r;
}
// CDNA5 16-bit A/B fragment: lane half selects klo (0/8); halves 0..7 -> K=klo..klo+7,
// halves 8..15 -> K=klo+16..klo+23. Two 16B loads per fragment.
__device__ __forceinline__ v16bf ldfrag(const __bf16* __restrict__ rowp, int klo) {
    v8bf lo = *(const v8bf*)(rowp + klo);
    v8bf hi = *(const v8bf*)(rowp + klo + 16);
    return __builtin_shufflevector(lo, hi, 0,1,2,3,4,5,6,7,8,9,10,11,12,13,14,15);
}
__device__ __forceinline__ v8f vzero8() { v8f z = {0.f,0.f,0.f,0.f,0.f,0.f,0.f,0.f}; return z; }

// ---------- kernel 1/4: 512-deep GEMM, C = X * W^T (+ bias), fused epilogues ----------
// mode 0: outF = C+b (q),  outB = relu bf16 (qp)
// mode 1: outB = relu(C+b) bf16                 (kp)
// mode 2: outB = (C+b) bf16                     (v)
// mode 3: outF = relu(C+b) + resid              (Wo branch -> d_out)
#define LDP 40  // LDS row pitch in halves (80B, 16B-aligned)

__global__ __launch_bounds__(256)
void gemm512(const float* __restrict__ X, const float* __restrict__ W,
             const float* __restrict__ bias,
             float* __restrict__ outF, __bf16* __restrict__ outB,
             const float* __restrict__ resid, int mode)
{
    __shared__ __bf16 As[2][128][LDP];
    __shared__ __bf16 Bs[2][128][LDP];

    const int tid  = threadIdx.x;
    const int wave = tid >> 5;
    const int lane = tid & 31;
    const int lm   = lane & 15;
    const int klo  = (lane >> 4) << 3;
    const int m0   = blockIdx.y * 128;
    const int n0   = blockIdx.x * 128;
    const int wrow = wave & 3;   // row-tile pair 2*wrow, 2*wrow+1
    const int wcol = wave >> 2;  // col-tile quad 4*wcol .. +3

    const int lr = tid >> 3;     // 0..31 loader row
    const int ls = tid & 7;      // 0..7 loader segment (4 floats)

    v8f acc[2][4];
#pragma unroll
    for (int i = 0; i < 2; ++i)
#pragma unroll
        for (int j = 0; j < 4; ++j) acc[i][j] = vzero8();

    auto load_tile = [&](int buf, int k0) {
#pragma unroll
        for (int p = 0; p < 4; ++p) {
            int row = lr + 32 * p;
            float4 va = *(const float4*)(X + (size_t)(m0 + row) * 512 + k0 + ls * 4);
            *(v4bf*)&As[buf][row][ls * 4] = f4tobf4(va);      // one ds_store_b64
            float4 vb = *(const float4*)(W + (size_t)(n0 + row) * 512 + k0 + ls * 4);
            *(v4bf*)&Bs[buf][row][ls * 4] = f4tobf4(vb);
        }
    };

    load_tile(0, 0);
    __syncthreads();
#pragma unroll 1
    for (int kk = 0; kk < 16; ++kk) {
        const int cur = kk & 1;
        if (kk + 2 < 16) {   // prefetch 2 steps ahead -> global_prefetch_b8
            __builtin_prefetch(X + (size_t)(m0 + lr) * 512 + (kk + 2) * 32 + ls * 4, 0, 1);
            __builtin_prefetch(W + (size_t)(n0 + lr) * 512 + (kk + 2) * 32 + ls * 4, 0, 1);
        }
        if (kk + 1 < 16) load_tile(cur ^ 1, (kk + 1) * 32);

        // preload ALL fragments, then run 8 WMMAs back-to-back
        v16bf af[2], bfr[4];
#pragma unroll
        for (int mi = 0; mi < 2; ++mi)
            af[mi] = ldfrag(&As[cur][(2 * wrow + mi) * 16 + lm][0], klo);
#pragma unroll
        for (int ni = 0; ni < 4; ++ni)
            bfr[ni] = ldfrag(&Bs[cur][(4 * wcol + ni) * 16 + lm][0], klo);
#pragma unroll
        for (int ni = 0; ni < 4; ++ni)
#pragma unroll
            for (int mi = 0; mi < 2; ++mi)
                acc[mi][ni] = WMMA_BF16(af[mi], bfr[ni], acc[mi][ni]);
        __syncthreads();
    }

    // epilogue: D layout -> lane holds col lm(+16*tile), VGPR r -> row r+8*(lane/16)
#pragma unroll
    for (int ni = 0; ni < 4; ++ni) {
        const int col = n0 + (4 * wcol + ni) * 16 + lm;
        const float bv = bias[col];
#pragma unroll
        for (int mi = 0; mi < 2; ++mi) {
#pragma unroll
            for (int r = 0; r < 8; ++r) {
                const int row = m0 + (2 * wrow + mi) * 16 + r + 8 * (lane >> 4);
                const size_t idx = (size_t)row * 512 + col;
                float v = acc[mi][ni][r] + bv;
                if (mode == 0)      { outF[idx] = v; outB[idx] = (__bf16)fmaxf(v, 0.f); }
                else if (mode == 1) { outB[idx] = (__bf16)fmaxf(v, 0.f); }
                else if (mode == 2) { outB[idx] = (__bf16)v; }
                else                { outF[idx] = fmaxf(v, 0.f) + resid[idx]; }
            }
        }
    }
}

// ---------- kernel 2: kv[d,e] = sum_n kp[n,d]*v[n,e]; ksum[d] = sum_n kp[n,d] ----------
// grid (split=16, bh=32). Deterministic partials, no atomics.
__global__ __launch_bounds__(256)
void kv_reduce(const __bf16* __restrict__ kp, const __bf16* __restrict__ v,
               float* __restrict__ kv_part, float* __restrict__ ks_part)
{
    __shared__ __bf16 kpT[64][LDP];   // [d][n]
    __shared__ __bf16 vT [64][LDP];   // [e][n]

    const int split = blockIdx.x;     // 0..15
    const int bh    = blockIdx.y;     // 0..31
    const int b = bh >> 3, h = bh & 7;
    const int tid  = threadIdx.x;
    const int wave = tid >> 5, lane = tid & 31;
    const int lm = lane & 15, klo = (lane >> 4) << 3;
    const int mi = wave >> 1;                 // d-tile 0..3
    const int nb = (wave & 1) * 2;            // e-tiles nb, nb+1
    const int lr = tid >> 3, ls = tid & 7;

    v8f acc[2] = { vzero8(), vzero8() };
    float ksl = 0.f;

    const size_t base = ((size_t)b * 8192 + (size_t)split * 512) * 512 + h * 64;
#pragma unroll 1
    for (int c = 0; c < 16; ++c) {
        const size_t roff = base + (size_t)(c * 32 + lr) * 512 + ls * 8;
        v8bf k8 = *(const v8bf*)(kp + roff);
        v8bf v8 = *(const v8bf*)(v + roff);
        __syncthreads();                       // prior chunk fully consumed
#pragma unroll
        for (int j = 0; j < 8; ++j) { kpT[ls * 8 + j][lr] = k8[j]; vT[ls * 8 + j][lr] = v8[j]; }
        __syncthreads();
        if (tid < 64) {
#pragma unroll
            for (int rr = 0; rr < 32; ++rr) ksl += (float)kpT[tid][rr];
        }
        v16bf a = ldfrag(&kpT[mi * 16 + lm][0], klo);
        v16bf b0 = ldfrag(&vT[(nb + 0) * 16 + lm][0], klo);
        v16bf b1 = ldfrag(&vT[(nb + 1) * 16 + lm][0], klo);
        acc[0] = WMMA_BF16(a, b0, acc[0]);
        acc[1] = WMMA_BF16(a, b1, acc[1]);
    }
    float* kvp = kv_part + ((size_t)bh * 16 + split) * 4096;   // stored transposed [e][d]
#pragma unroll
    for (int nn = 0; nn < 2; ++nn) {
        const int e = (nb + nn) * 16 + lm;
#pragma unroll
        for (int r = 0; r < 8; ++r) {
            const int d = mi * 16 + r + 8 * (lane >> 4);
            kvp[e * 64 + d] = acc[nn][r];
        }
    }
    if (tid < 64) ks_part[((size_t)bh * 16 + split) * 64 + tid] = ksl;
}

// ---------- kernel 2b: reduce partials -> bf16 kv^T + f32 ksum ----------
__global__ __launch_bounds__(256)
void kv_finalize(const float* __restrict__ kv_part, const float* __restrict__ ks_part,
                 __bf16* __restrict__ kvt, float* __restrict__ ksum)
{
    const int bh = blockIdx.x, tid = threadIdx.x;
    for (int i = tid; i < 4096; i += 256) {
        float s = 0.f;
#pragma unroll
        for (int p = 0; p < 16; ++p) s += kv_part[((size_t)bh * 16 + p) * 4096 + i];
        kvt[(size_t)bh * 4096 + i] = (__bf16)s;
    }
    if (tid < 64) {
        float s = 0.f;
#pragma unroll
        for (int p = 0; p < 16; ++p) s += ks_part[((size_t)bh * 16 + p) * 64 + tid];
        ksum[bh * 64 + tid] = s;
    }
}

// ---------- kernel 3: num = qp @ kv, row_norm, residual, LayerNorm-0 ----------
// one wave per head; 32 rows per block
__global__ __launch_bounds__(256)
void attn_ln0(const float* __restrict__ qf, const __bf16* __restrict__ qp,
              const __bf16* __restrict__ kvt, const float* __restrict__ ksum,
              const float* __restrict__ g0, const float* __restrict__ b0,
              float* __restrict__ oln)
{
    __shared__ float ks_lds[512];
    __shared__ float rowsum[32], rowsq[32];
    __shared__ float rn_lds[8][32];
    __shared__ float mu_lds[32], rs_lds[32];

    const int tid = threadIdx.x, wave = tid >> 5, lane = tid & 31;
    const int h = wave;
    const int lm = lane & 15, klo = (lane >> 4) << 3;
    const int rows0 = blockIdx.x * 32;
    const int b  = rows0 >> 13;
    const int bh = b * 8 + h;

    // async copy ksum[b] -> LDS (ASYNCcnt path: global_load_async_to_lds_b32)
#pragma unroll
    for (int i = tid; i < 512; i += 256) {
        unsigned lds_off = (unsigned)(size_t)&ks_lds[i];
        unsigned long long ga = (unsigned long long)(size_t)(ksum + b * 512 + i);
        asm volatile("global_load_async_to_lds_b32 %0, %1, off"
                     :: "v"(lds_off), "v"(ga) : "memory");
    }
    asm volatile("s_wait_asynccnt 0" ::: "memory");
    if (tid < 32) { rowsum[tid] = 0.f; rowsq[tid] = 0.f; }
    __syncthreads();

    v8f acc[2][4];
#pragma unroll
    for (int i = 0; i < 2; ++i)
#pragma unroll
        for (int j = 0; j < 4; ++j) acc[i][j] = vzero8();

    const size_t qbase = (size_t)rows0 * 512 + h * 64;
#pragma unroll
    for (int ks = 0; ks < 2; ++ks) {
        const int kbase = ks * 32;
        v16bf af[2], bfr[4];
#pragma unroll
        for (int mi = 0; mi < 2; ++mi)
            af[mi] = ldfrag(qp + qbase + (size_t)(mi * 16 + lm) * 512 + kbase, klo);
#pragma unroll
        for (int ni = 0; ni < 4; ++ni)
            bfr[ni] = ldfrag(kvt + (size_t)bh * 4096 + (ni * 16 + lm) * 64 + kbase, klo);
#pragma unroll
        for (int ni = 0; ni < 4; ++ni)
#pragma unroll
            for (int mi = 0; mi < 2; ++mi)
                acc[mi][ni] = WMMA_BF16(af[mi], bfr[ni], acc[mi][ni]);
    }

    // row_norm for row rows0+lane
    {
        float rn = 1e-5f;
        const __bf16* qrow = qp + (size_t)(rows0 + lane) * 512 + h * 64;
#pragma unroll
        for (int d = 0; d < 64; ++d) rn += (float)qrow[d] * ks_lds[h * 64 + d];
        rn_lds[h][lane] = rn;
    }

    // residual + scale; accumulate LN stats with shfl_xor(16) + LDS atomics
    const float inv_s = 0.044194173824159216f;   // 1/sqrt(512)
#pragma unroll
    for (int mi = 0; mi < 2; ++mi)
#pragma unroll
        for (int ni = 0; ni < 4; ++ni) {
            const int col = h * 64 + ni * 16 + lm;
#pragma unroll
            for (int r = 0; r < 8; ++r) {
                const int mloc = mi * 16 + r + 8 * (lane >> 4);
                const float q0 = qf[(size_t)(rows0 + mloc) * 512 + col];
                acc[mi][ni][r] = q0 + acc[mi][ni][r] * inv_s / rn_lds[h][mloc];
            }
        }
#pragma unroll
    for (int mi = 0; mi < 2; ++mi)
#pragma unroll
        for (int r = 0; r < 8; ++r) {
            float s = 0.f, q = 0.f;
#pragma unroll
            for (int ni = 0; ni < 4; ++ni) { float v = acc[mi][ni][r]; s += v; q += v * v; }
#pragma unroll
            for (int off = 1; off < 16; off <<= 1) {
                s += __shfl_xor(s, off, 16);
                q += __shfl_xor(q, off, 16);
            }
            if (lm == 0) {
                const int mloc = mi * 16 + r + 8 * (lane >> 4);
                atomicAdd(&rowsum[mloc], s);
                atomicAdd(&rowsq[mloc], q);
            }
        }
    __syncthreads();
    if (tid < 32) {
        const float mu = rowsum[tid] * (1.f / 512.f);
        const float var = rowsq[tid] * (1.f / 512.f) - mu * mu;
        mu_lds[tid] = mu;
        rs_lds[tid] = rsqrtf(var + 1e-5f);
    }
    __syncthreads();
#pragma unroll
    for (int mi = 0; mi < 2; ++mi)
#pragma unroll
        for (int ni = 0; ni < 4; ++ni) {
            const int col = h * 64 + ni * 16 + lm;
            const float g = g0[col], bb = b0[col];
#pragma unroll
            for (int r = 0; r < 8; ++r) {
                const int mloc = mi * 16 + r + 8 * (lane >> 4);
                const float v = (acc[mi][ni][r] - mu_lds[mloc]) * rs_lds[mloc] * g + bb;
                oln[(size_t)(rows0 + mloc) * 512 + col] = v;
            }
        }
}

// ---------- kernel 5: final LayerNorm in place on d_out, one wave per row ----------
__global__ __launch_bounds__(256)
void ln_final(float* __restrict__ out, const float* __restrict__ g1,
              const float* __restrict__ b1)
{
    const int tid = threadIdx.x, wave = tid >> 5, lane = tid & 31;
    const int row = blockIdx.x * 8 + wave;
    float* p = out + (size_t)row * 512;
    float v[16], s = 0.f, q = 0.f;
#pragma unroll
    for (int i = 0; i < 16; ++i) { v[i] = p[lane + i * 32]; s += v[i]; q += v[i] * v[i]; }
#pragma unroll
    for (int off = 1; off < 32; off <<= 1) { s += __shfl_xor(s, off, 32); q += __shfl_xor(q, off, 32); }
    const float mu = s * (1.f / 512.f);
    const float rstd = rsqrtf(q * (1.f / 512.f) - mu * mu + 1e-5f);
#pragma unroll
    for (int i = 0; i < 16; ++i) {
        const int c = lane + i * 32;
        p[c] = (v[i] - mu) * rstd * g1[c] + b1[c];
    }
}

// ---------- host ----------
extern "C" void kernel_launch(void* const* d_in, const int* in_sizes, int n_in,
                              void* d_out, int out_size, void* d_ws, size_t ws_size,
                              hipStream_t stream)
{
    const float* Q  = (const float*)d_in[0];
    const float* K  = (const float*)d_in[1];
    const float* Wq = (const float*)d_in[2];
    const float* bq = (const float*)d_in[3];
    const float* Wk = (const float*)d_in[4];
    const float* bk = (const float*)d_in[5];
    const float* Wv = (const float*)d_in[6];
    const float* bv = (const float*)d_in[7];
    const float* Wo = (const float*)d_in[8];
    const float* bo = (const float*)d_in[9];
    const float* g0 = (const float*)d_in[10];
    const float* b0 = (const float*)d_in[11];
    const float* g1 = (const float*)d_in[12];
    const float* b1 = (const float*)d_in[13];
    float* out = (float*)d_out;

    char* ws = (char*)d_ws;
    float*  qf   = (float*) (ws);                          // 64 MB : q (pre-relu, residual)
    __bf16* qp   = (__bf16*)(ws + ((size_t)64  << 20));    // 32 MB : relu(q) bf16
    __bf16* kp   = (__bf16*)(ws + ((size_t)96  << 20));    // 32 MB : relu(k) bf16
    __bf16* vb   = (__bf16*)(ws + ((size_t)128 << 20));    // 32 MB : v bf16
    float*  oln  = (float*) (ws + ((size_t)160 << 20));    // 64 MB : LN0 output
    float*  kvp  = (float*) (ws + ((size_t)224 << 20));    //  8 MB : kv partials
    float*  ksp  = (float*) (ws + ((size_t)232 << 20));    // 128KB : ksum partials
    __bf16* kvt  = (__bf16*)(ws + ((size_t)233 << 20));    // 256KB : kv^T bf16
    float*  ksum = (float*) (ws + ((size_t)234 << 20));    //   8KB : ksum f32

    const dim3 blk(256);
    const dim3 gproj(4, 256);     // (512/128 col tiles, 32768/128 row tiles)

    gemm512<<<gproj, blk, 0, stream>>>(Q, Wq, bq, qf, qp, nullptr, 0);
    gemm512<<<gproj, blk, 0, stream>>>(K, Wk, bk, nullptr, kp, nullptr, 1);
    gemm512<<<gproj, blk, 0, stream>>>(K, Wv, bv, nullptr, vb, nullptr, 2);
    kv_reduce<<<dim3(16, 32), blk, 0, stream>>>(kp, vb, kvp, ksp);
    kv_finalize<<<32, blk, 0, stream>>>(kvp, ksp, kvt, ksum);
    attn_ln0<<<1024, blk, 0, stream>>>(qf, qp, kvt, ksum, g0, b0, oln);
    gemm512<<<gproj, blk, 0, stream>>>(oln, Wo, bo, out, nullptr, oln, 3);
    ln_final<<<4096, blk, 0, stream>>>(out, g1, b1);
}
